// DGCNN_cls_28810640621725
// MI455X (gfx1250) — matrix operand order repaired and checked by reading, not compiled, not run on hardware
//
#include <hip/hip_runtime.h>
#include <hip/hip_bf16.h>

typedef __attribute__((ext_vector_type(16))) _Float16 v16h;
typedef __attribute__((ext_vector_type(8)))  _Float16 v8h;
typedef __attribute__((ext_vector_type(8)))  float    v8f;

#define LRELU_NEG 0.2f
#define NPTS 2048
#define NB   8

__device__ __forceinline__ float lrelu(float x) { return x > 0.f ? x : LRELU_NEG * x; }

// ---------------- WMMA fragment loaders (gfx1250 wave32, v_wmma_f32_16x16x32_f16) ----------
// A fragment 16x32 (MxK) from row-major A (leading dim lda):
//   lanes 0-15 : m=lane,    k in {0..7} then {16..23}
//   lanes 16-31: m=lane-16, k in {8..15} then {24..31}
// All leading dims are multiples of 16 halfs and bases 256B-aligned -> 16B vector loads OK.
__device__ __forceinline__ v16h load_a_frag(const _Float16* __restrict__ A, int lda,
                                            int row, int k0, int lane) {
  const _Float16* p = A + (size_t)row * lda + k0 + ((lane >> 4) * 8);
  v8h lo = *(const v8h*)p;          // k+0..7   (global_load_b128)
  v8h hi = *(const v8h*)(p + 16);   // k+16..23 (global_load_b128)
  v16h a;
#pragma unroll
  for (int i = 0; i < 8; ++i) { a[i] = lo[i]; a[8 + i] = hi[i]; }
  return a;
}
// B fragment 32x16 (KxN) where B[k][n] = Bt[n][k], Bt row-major (leading dim ldb):
//   lanes 0-15 : n=lane,    k in {0..7} then {16..23}
//   lanes 16-31: n=lane-16, k in {8..15} then {24..31}
// Out-of-range columns are clamped to row 0 by the caller; no zeroing needed because WMMA
// column n of D depends only on column n of B, and OOB lanes are masked in the epilogue.
__device__ __forceinline__ v16h load_b_frag(const _Float16* __restrict__ Bt, int ldb,
                                            int col, int k0, int lane) {
  const _Float16* p = Bt + (size_t)col * ldb + k0 + ((lane >> 4) * 8);
  v8h lo = *(const v8h*)p;          // global_load_b128
  v8h hi = *(const v8h*)(p + 16);   // global_load_b128
  v16h b;
#pragma unroll
  for (int i = 0; i < 8; ++i) { b[i] = lo[i]; b[8 + i] = hi[i]; }
  return b;
}

// ---------------- generic NT GEMM: C[m,n] = epi( sum_k A[m,k] * Bt[n,k] ) ------------------
// block = 256 threads = 8 waves, each wave computes one 16x16 tile via WMMA.
// epilogue: v = (acc + bias[n]) * sc[n] + bi[n]; optional LeakyReLU; f32 and/or f16 stores.
__global__ void __launch_bounds__(256)
gemm_nt(const _Float16* __restrict__ A, int lda,
        const _Float16* __restrict__ Bt, int ldb,
        int Kp, int Mreal, int Nreal,
        float* __restrict__ Cout, int ldc,
        _Float16* __restrict__ Hout, int ldh,
        const float* __restrict__ sc, const float* __restrict__ bi,
        const float* __restrict__ bias, int do_act)
{
  const int lane = threadIdx.x & 31;
  const int wave = threadIdx.x >> 5;
  const int m0 = (blockIdx.y * 8 + wave) * 16;
  const int n0 = blockIdx.x * 16;
  int am = m0 + (lane & 15); if (am >= Mreal) am = Mreal - 1;   // clamp; masked at store
  const int bn = n0 + (lane & 15);
  const bool bvalid = bn < Nreal;
  const int bnc = bvalid ? bn : 0;                              // clamp; masked at store
  v8f acc = {0.f, 0.f, 0.f, 0.f, 0.f, 0.f, 0.f, 0.f};
  for (int k0 = 0; k0 < Kp; k0 += 32) {
    v16h a = load_a_frag(A, lda, am, k0, lane);
    v16h b = load_b_frag(Bt, ldb, bnc, k0, lane);
    acc = __builtin_amdgcn_wmma_f32_16x16x32_f16(false, a, false, b, (short)0, acc,
                                                 false, false);
  }
  if (!bvalid) return;
  const float s_ = sc ? sc[bn] : 1.f;
  const float b_ = bi ? bi[bn] : 0.f;
  const float bs = bias ? bias[bn] : 0.f;
  const int mbase = m0 + ((lane >> 4) << 3);
#pragma unroll
  for (int e = 0; e < 8; ++e) {
    int m = mbase + e;
    if (m < Mreal) {
      float v = (acc[e] + bs) * s_ + b_;
      if (do_act) v = lrelu(v);
      if (Cout) Cout[(size_t)m * ldc + bn] = v;
      if (Hout) Hout[(size_t)m * ldh + bn] = (_Float16)v;
    }
  }
}

// ---------------- KNN: pairwise distance d = 2*X.X^T - xx_r - xx_c (one batch) -------------
__global__ void __launch_bounds__(256)
knn_dist(const _Float16* __restrict__ X, int lda, int Kp,
         const float* __restrict__ xx, float* __restrict__ D)
{
  const int lane = threadIdx.x & 31;
  const int wave = threadIdx.x >> 5;
  const int m0 = (blockIdx.y * 8 + wave) * 16;
  const int n0 = blockIdx.x * 16;
  const int am = m0 + (lane & 15);
  const int bn = n0 + (lane & 15);
  v8f acc = {0.f, 0.f, 0.f, 0.f, 0.f, 0.f, 0.f, 0.f};
  for (int k0 = 0; k0 < Kp; k0 += 32) {
    v16h a = load_a_frag(X, lda, am, k0, lane);
    v16h b = load_b_frag(X, lda, bn, k0, lane);
    acc = __builtin_amdgcn_wmma_f32_16x16x32_f16(false, a, false, b, (short)0, acc,
                                                 false, false);
  }
  const float xxn = xx[bn];
  const int mbase = m0 + ((lane >> 4) << 3);
#pragma unroll
  for (int e = 0; e < 8; ++e) {
    int m = mbase + e;
    D[(size_t)m * NPTS + bn] = 2.f * acc[e] - xx[m] - xxn;
  }
}

// ---------------- KNN: top-32 per row (descending, ties -> lower index) --------------------
// One wave per row; row held in 64 VGPRs/lane; removal via static-index skip bitmask.
__global__ void __launch_bounds__(256)
knn_topk(const float* __restrict__ D, int* __restrict__ idxout)
{
  const int lane = threadIdx.x & 31;
  const int wave = threadIdx.x >> 5;
  const int row = blockIdx.x * 8 + wave;
  const float* drow = D + (size_t)row * NPTS;
  float v[64];
#pragma unroll
  for (int i = 0; i < 64; ++i) v[i] = drow[lane + 32 * i];
  unsigned long long removed = 0ull;
  float bv = v[0]; int bidx = 0;
#pragma unroll
  for (int i = 1; i < 64; ++i) if (v[i] > bv) { bv = v[i]; bidx = i; }
  for (int kk = 0; kk < 32; ++kk) {
    float mv = bv;
    int mc = lane + 32 * bidx;
#pragma unroll
    for (int offm = 16; offm > 0; offm >>= 1) {
      float ov = __shfl_xor(mv, offm, 32);
      int oc = __shfl_xor(mc, offm, 32);
      if (ov > mv || (ov == mv && oc < mc)) { mv = ov; mc = oc; }
    }
    if (lane == 0) idxout[row * 32 + kk] = mc;
    if ((mc & 31) == lane) {          // winner lane removes element and rescans
      removed |= 1ull << (mc >> 5);
      bv = -3.0e38f; bidx = 0;
#pragma unroll
      for (int i = 0; i < 64; ++i) {
        bool ok = ((removed >> i) & 1ull) == 0ull;
        if (ok && v[i] > bv) { bv = v[i]; bidx = i; }
      }
    }
  }
}

// ---------------- EdgeConv epilogue: out[n,o] = max_k lrelu((U[idx[k],o]+T[n,o])*s+b) ------
// UT: [B*N, 2O], cols [0,O)=U, [O,2O)=T. Writes f16 feature (into cat buffer) + sumsq (xx).
__global__ void edge_gather_max(const float* __restrict__ UT, int O,
                                const int* __restrict__ idx,
                                const float* __restrict__ sc, const float* __restrict__ bi,
                                _Float16* __restrict__ outH, int ldh,
                                float* __restrict__ xxout)
{
  const int b = blockIdx.y, n = blockIdx.x;
  const size_t row = (size_t)b * NPTS + n;
  __shared__ int sidx[32];
  __shared__ float red[256];
  if (threadIdx.x < 32) sidx[threadIdx.x] = idx[row * 32 + threadIdx.x];
  __syncthreads();
  const int o = threadIdx.x;            // blockDim.x == O (64/128/256)
  const int ld = 2 * O;
  const float t = UT[row * (size_t)ld + O + o];
  const float s_ = sc[o], b_ = bi[o];
  const float* Ub = UT + (size_t)b * NPTS * ld + o;
  float m = -3.0e38f;
#pragma unroll 8
  for (int k = 0; k < 32; ++k) {
    float u = Ub[(size_t)sidx[k] * ld];
    m = fmaxf(m, lrelu((u + t) * s_ + b_));
  }
  _Float16 mh = (_Float16)m;
  outH[row * (size_t)ldh + o] = mh;
  if (xxout) {
    float mv = (float)mh;
    red[o] = mv * mv;
    __syncthreads();
    for (int s = blockDim.x >> 1; s > 0; s >>= 1) {
      if (o < s) red[o] += red[o + s];
      __syncthreads();
    }
    if (o == 0) xxout[row] = red[0];
  }
}

// ---------------- pooling over N: max (y branch) and [max, mean] (g branch), f16 out -------
__global__ void pool_hw(const float* __restrict__ H,
                        _Float16* __restrict__ ymaxH, _Float16* __restrict__ gH)
{
  const int o = blockIdx.x * blockDim.x + threadIdx.x;   // 0..1023
  const int b = blockIdx.y;                              // 0..7
  const float* col = H + (size_t)b * NPTS * 1024 + o;
  float mx = -3.0e38f, sm = 0.f;
  for (int n = 0; n < NPTS; ++n) {
    float v = col[(size_t)n * 1024];
    mx = fmaxf(mx, v); sm += v;
  }
  ymaxH[b * 1024 + o] = (_Float16)mx;
  gH[b * 2048 + o] = (_Float16)mx;
  gH[b * 2048 + 1024 + o] = (_Float16)(sm * (1.f / NPTS));
}

// ---------------- preps ----------------
__global__ void prep_pts(const float* __restrict__ x,
                         _Float16* __restrict__ ptsH, float* __restrict__ xx0)
{
  const int g = blockIdx.x * blockDim.x + threadIdx.x;   // b*N + n
  if (g >= NB * NPTS) return;
  const int b = g / NPTS, n = g % NPTS;
  float ss = 0.f;
#pragma unroll
  for (int c = 0; c < 3; ++c) {
    float v = x[((size_t)b * 3 + c) * NPTS + n];
    _Float16 h = (_Float16)v;
    ptsH[(size_t)g * 32 + c] = h;
    float hv = (float)h;
    ss += hv * hv;
  }
#pragma unroll
  for (int c = 3; c < 32; ++c) ptsH[(size_t)g * 32 + c] = (_Float16)0.f;
  xx0[g] = ss;
}

// Wcat[o][k] (row-major [2O, Kp] f16):  o<O -> W[o,k];  o>=O -> W[o-O, C+k] - W[o-O, k]; pad 0
__global__ void prep_wcat(const float* __restrict__ W, int C, int Kp, int O,
                          _Float16* __restrict__ Wcat)
{
  const int k = blockIdx.x * blockDim.x + threadIdx.x;
  const int o = blockIdx.y;            // 0..2O-1
  if (k >= Kp) return;
  float v = 0.f;
  if (k < C) {
    if (o < O) v = W[(size_t)o * 2 * C + k];
    else       v = W[(size_t)(o - O) * 2 * C + C + k] - W[(size_t)(o - O) * 2 * C + k];
  }
  Wcat[(size_t)o * Kp + k] = (_Float16)v;
}

__global__ void f32_to_f16(const float* __restrict__ src, _Float16* __restrict__ dst, int n)
{
  const int i = blockIdx.x * blockDim.x + threadIdx.x;
  if (i < n) dst[i] = (_Float16)src[i];
}

// ==========================================================================================
extern "C" void kernel_launch(void* const* d_in, const int* in_sizes, int n_in,
                              void* d_out, int out_size, void* d_ws, size_t ws_size,
                              hipStream_t stream) {
  (void)in_sizes; (void)n_in; (void)out_size; (void)ws_size;
  const float* x   = (const float*)d_in[0];
  const float* W1  = (const float*)d_in[1];  const float* s1 = (const float*)d_in[2];  const float* b1 = (const float*)d_in[3];
  const float* W2  = (const float*)d_in[4];  const float* s2 = (const float*)d_in[5];  const float* b2 = (const float*)d_in[6];
  const float* W3  = (const float*)d_in[7];  const float* s3 = (const float*)d_in[8];  const float* b3 = (const float*)d_in[9];
  const float* W4  = (const float*)d_in[10]; const float* s4 = (const float*)d_in[11]; const float* b4 = (const float*)d_in[12];
  const float* W5  = (const float*)d_in[13]; const float* s5 = (const float*)d_in[14]; const float* b5 = (const float*)d_in[15];
  const float* L1w = (const float*)d_in[16]; const float* s6 = (const float*)d_in[17]; const float* b6 = (const float*)d_in[18];
  const float* L2w = (const float*)d_in[19]; const float* L2b = (const float*)d_in[20];
  const float* s7  = (const float*)d_in[21]; const float* b7 = (const float*)d_in[22];
  const float* L3w = (const float*)d_in[23]; const float* L3b = (const float*)d_in[24];
  const float* F1w = (const float*)d_in[25]; const float* s8 = (const float*)d_in[26]; const float* b8 = (const float*)d_in[27];
  const float* F2w = (const float*)d_in[28]; const float* F2b = (const float*)d_in[29];
  const float* s9  = (const float*)d_in[30]; const float* b9 = (const float*)d_in[31];
  const float* F3w = (const float*)d_in[32]; const float* F3b = (const float*)d_in[33];
  float* out = (float*)d_out;                 // [0,40) = g, [40,80) = y

  const int M = NB * NPTS;                    // 16384 rows (all batches stacked)
  size_t off = 0;
  auto alloc = [&](size_t bytes) -> char* {
    off = (off + 255) & ~(size_t)255;
    char* p = (char*)d_ws + off;
    off += bytes;
    return p;
  };
  // ~110 MB of workspace total
  _Float16* ptsH = (_Float16*)alloc((size_t)M * 32 * 2);     // pts padded to K=32, f16
  float* xx0 = (float*)alloc((size_t)M * 4);
  float* xx1 = (float*)alloc((size_t)M * 4);
  float* xx2 = (float*)alloc((size_t)M * 4);
  float* xx3 = (float*)alloc((size_t)M * 4);
  _Float16* catH = (_Float16*)alloc((size_t)M * 512 * 2);    // [x1|x2|x3|x4] f16, ld=512
  int* idxb  = (int*)alloc((size_t)M * 32 * 4);
  float* dist = (float*)alloc((size_t)NPTS * NPTS * 4);      // per-batch scratch (reused)
  float* hbuf = (float*)alloc((size_t)M * 1024 * 4);         // UT scratch, then h
  _Float16* Wc1 = (_Float16*)alloc(128 * 32 * 2);
  _Float16* Wc2 = (_Float16*)alloc(128 * 64 * 2);
  _Float16* Wc3 = (_Float16*)alloc(256 * 64 * 2);
  _Float16* Wc4 = (_Float16*)alloc(512 * 128 * 2);
  _Float16* W5h = (_Float16*)alloc(1024 * 512 * 2);
  _Float16* L1h = (_Float16*)alloc(512 * 2048 * 2);
  _Float16* L2h = (_Float16*)alloc(256 * 512 * 2);
  _Float16* L3h = (_Float16*)alloc(5 * 256 * 2);
  _Float16* F1h = (_Float16*)alloc(512 * 1024 * 2);
  _Float16* F2h = (_Float16*)alloc(256 * 512 * 2);
  _Float16* F3h = (_Float16*)alloc(5 * 256 * 2);
  _Float16* ymaxH = (_Float16*)alloc(16 * 1024 * 2);
  _Float16* gpH   = (_Float16*)alloc(16 * 2048 * 2);
  _Float16* y1H = (_Float16*)alloc(16 * 512 * 2);
  _Float16* y2H = (_Float16*)alloc(16 * 256 * 2);
  _Float16* g1H = (_Float16*)alloc(16 * 512 * 2);
  _Float16* g2H = (_Float16*)alloc(16 * 256 * 2);

  // ---- weight preparation (f32 -> f16; EdgeConv weights folded: [W1 ; W2-W1]) ----
  f32_to_f16<<<(1024 * 512 + 255) / 256, 256, 0, stream>>>(W5, W5h, 1024 * 512);
  f32_to_f16<<<(512 * 2048 + 255) / 256, 256, 0, stream>>>(L1w, L1h, 512 * 2048);
  f32_to_f16<<<(256 * 512 + 255) / 256, 256, 0, stream>>>(L2w, L2h, 256 * 512);
  f32_to_f16<<<(5 * 256 + 255) / 256, 256, 0, stream>>>(L3w, L3h, 5 * 256);
  f32_to_f16<<<(512 * 1024 + 255) / 256, 256, 0, stream>>>(F1w, F1h, 512 * 1024);
  f32_to_f16<<<(256 * 512 + 255) / 256, 256, 0, stream>>>(F2w, F2h, 256 * 512);
  f32_to_f16<<<(5 * 256 + 255) / 256, 256, 0, stream>>>(F3w, F3h, 5 * 256);
  prep_wcat<<<dim3(1, 128), 64, 0, stream>>>(W1, 3, 32, 64, Wc1);
  prep_wcat<<<dim3(1, 128), 64, 0, stream>>>(W2, 64, 64, 64, Wc2);
  prep_wcat<<<dim3(1, 256), 64, 0, stream>>>(W3, 64, 64, 128, Wc3);
  prep_wcat<<<dim3(2, 512), 64, 0, stream>>>(W4, 128, 128, 256, Wc4);
  prep_pts<<<(M + 255) / 256, 256, 0, stream>>>(x, ptsH, xx0);

  // ---- four EdgeConv layers ----
  struct Lcfg {
    const _Float16* feat; int lda, Kp; const float* xxin; const _Float16* wc; int O;
    const float* sc; const float* bi; _Float16* outH; float* xxout;
  };
  Lcfg ls[4] = {
    { ptsH,       32,  32,  xx0, Wc1, 64,  s1, b1, catH + 0,   xx1 },
    { catH + 0,   512, 64,  xx1, Wc2, 64,  s2, b2, catH + 64,  xx2 },
    { catH + 64,  512, 64,  xx2, Wc3, 128, s3, b3, catH + 128, xx3 },
    { catH + 128, 512, 128, xx3, Wc4, 256, s4, b4, catH + 256, nullptr },
  };
  for (int l = 0; l < 4; ++l) {
    const Lcfg& P = ls[l];
    for (int b = 0; b < NB; ++b) {
      knn_dist<<<dim3(NPTS / 16, NPTS / 128), 256, 0, stream>>>(
          P.feat + (size_t)b * NPTS * P.lda, P.lda, P.Kp, P.xxin + (size_t)b * NPTS, dist);
      knn_topk<<<NPTS / 8, 256, 0, stream>>>(dist, idxb + (size_t)b * NPTS * 32);
    }
    const int O2 = 2 * P.O;
    gemm_nt<<<dim3(O2 / 16, M / 128), 256, 0, stream>>>(
        P.feat, P.lda, P.wc, P.Kp, P.Kp, M, O2, hbuf, O2, nullptr, 0,
        nullptr, nullptr, nullptr, 0);
    edge_gather_max<<<dim3(NPTS, NB), P.O, 0, stream>>>(
        hbuf, P.O, idxb, P.sc, P.bi, P.outH, 512, P.xxout);
  }

  // ---- h = lrelu((cat @ W5^T) * s5 + b5), then pooling ----
  gemm_nt<<<dim3(1024 / 16, M / 128), 256, 0, stream>>>(
      catH, 512, W5h, 512, 512, M, 1024, hbuf, 1024, nullptr, 0, s5, b5, nullptr, 1);
  pool_hw<<<dim3(4, NB), 256, 0, stream>>>(hbuf, ymaxH, gpH);

  // ---- g branch: [max,mean] -> L1 -> L2 -> L3 ----
  gemm_nt<<<dim3(32, 1), 256, 0, stream>>>(gpH, 2048, L1h, 2048, 2048, 8, 512,
                                           nullptr, 0, g1H, 512, s6, b6, nullptr, 1);
  gemm_nt<<<dim3(16, 1), 256, 0, stream>>>(g1H, 512, L2h, 512, 512, 8, 256,
                                           nullptr, 0, g2H, 256, s7, b7, L2b, 1);
  gemm_nt<<<dim3(1, 1), 256, 0, stream>>>(g2H, 256, L3h, 256, 256, 8, 5,
                                          out, 5, nullptr, 0, nullptr, nullptr, L3b, 0);
  // ---- y branch: max -> F1 -> F2 -> F3 ----
  gemm_nt<<<dim3(32, 1), 256, 0, stream>>>(ymaxH, 1024, F1h, 1024, 1024, 8, 512,
                                           nullptr, 0, y1H, 512, s8, b8, nullptr, 1);
  gemm_nt<<<dim3(16, 1), 256, 0, stream>>>(y1H, 512, F2h, 512, 512, 8, 256,
                                           nullptr, 0, y2H, 256, s9, b9, F2b, 1);
  gemm_nt<<<dim3(1, 1), 256, 0, stream>>>(y2H, 256, F3h, 256, 256, 8, 5,
                                          out + 40, 5, nullptr, 0, nullptr, nullptr, F3b, 0);
}
